// WIN_9268539425544
// MI455X (gfx1250) — matrix operand order repaired and checked by reading, not compiled, not run on hardware
//
#include <hip/hip_runtime.h>
#include <hip/hip_bf16.h>
#include <math.h>

// ---------------------------------------------------------------------------
// Problem constants (from reference): B=128, T=512 (body), P=256 (pun),
// EMB=300 (padded to 320), H=64, 4H=256, both directions packed -> 512 gates.
// ---------------------------------------------------------------------------
#define NB      128
#define TBODY   512
#define TPUN    256
#define EMB_K   300
#define KPAD    320
#define HH      64
#define G4      256          // 4*H
#define GATES   512          // fwd 4H | bwd 4H
#define DM      128          // 2*H (bilstm output dim)

typedef __bf16 bf16_t;
typedef __attribute__((ext_vector_type(16))) __bf16 v16bf;
typedef __attribute__((ext_vector_type(8)))  __bf16 bf16x8;
typedef __attribute__((ext_vector_type(2)))  __bf16 bf16x2;
typedef __attribute__((ext_vector_type(8)))  float  v8f;

__device__ __forceinline__ bf16_t f2bf(float f) {
  unsigned u = __builtin_bit_cast(unsigned, f);
  unsigned r = (u + 0x7FFFu + ((u >> 16) & 1u)) >> 16;   // round-nearest-even
  unsigned short s = (unsigned short)r;
  return __builtin_bit_cast(bf16_t, s);
}
__device__ __forceinline__ float bf2f(bf16_t b) {
  unsigned short s = __builtin_bit_cast(unsigned short, b);
  unsigned u = ((unsigned)s) << 16;
  return __builtin_bit_cast(float, u);
}
__device__ __forceinline__ float sigmoidf(float x) { return 1.0f / (1.0f + expf(-x)); }

// ---------------------------------------------------------------------------
// WMMA bf16 fragment loaders (ISA 7.12.2 layouts, wave32).
// A: 16x32 tile, row-major source A[m][k] (lda elems). Lane l: m=l&15,
//    khalf=l>>4; vgpr j: k = (j>>2)*16 + khalf*8 + (j&3)*2 (pair).
// B: 32x16 tile of B[k][n], sourced from an N-major buffer Bt[n][k] so each
//    lane (n=l&15, khalf=l>>4) reads k-pairs: k = (j>>2)*16 + khalf*8 + (j&3)*2.
// ---------------------------------------------------------------------------
__device__ __forceinline__ v16bf load_a_frag(const bf16_t* A, int lda, int lane) {
  int m = lane & 15, khalf = lane >> 4;
  v16bf a;
#pragma unroll
  for (int j = 0; j < 8; ++j) {
    int k = ((j >> 2) << 4) + (khalf << 3) + ((j & 3) << 1);
    bf16x2 p = *(const bf16x2*)(A + (size_t)m * lda + k);
    a[2 * j] = p[0]; a[2 * j + 1] = p[1];
  }
  return a;
}
__device__ __forceinline__ v16bf load_bt_frag(const bf16_t* Bt, int ldk, int lane) {
  int n = lane & 15, khalf = lane >> 4;
  v16bf b;
#pragma unroll
  for (int j = 0; j < 8; ++j) {
    int k = ((j >> 2) << 4) + (khalf << 3) + ((j & 3) << 1);
    bf16x2 p = *(const bf16x2*)(Bt + (size_t)n * ldk + k);
    b[2 * j] = p[0]; b[2 * j + 1] = p[1];
  }
  return b;
}
#define WMMA_BF16(a, b, c) \
  __builtin_amdgcn_wmma_f32_16x16x32_bf16(false, (a), false, (b), (short)0, (c), false, false)

// ---------------------------------------------------------------------------
// K0: sequence lengths (count of nonzero token ids per row)
// lens[0..127]=body, lens[128..255]=pun
// ---------------------------------------------------------------------------
__global__ __launch_bounds__(256) void seq_lens_kernel(const int* __restrict__ body,
                                                       const int* __restrict__ pun,
                                                       int* __restrict__ lens) {
  __shared__ int s[256];
  int b = blockIdx.x, isPun = blockIdx.y;
  int T = isPun ? TPUN : TBODY;
  const int* p = isPun ? (pun + (size_t)b * TPUN) : (body + (size_t)b * TBODY);
  int c = 0;
  for (int t = threadIdx.x; t < T; t += 256) c += (p[t] != 0) ? 1 : 0;
  s[threadIdx.x] = c; __syncthreads();
  for (int o = 128; o > 0; o >>= 1) {
    if (threadIdx.x < o) s[threadIdx.x] += s[threadIdx.x + o];
    __syncthreads();
  }
  if (threadIdx.x == 0) lens[isPun * NB + b] = s[0];
}

// ---------------------------------------------------------------------------
// K1: pack Wih (fwd|bwd) -> bf16 Wt[512][320], N-major, K zero-padded 300->320
// ---------------------------------------------------------------------------
__global__ __launch_bounds__(256) void pack_wih_kernel(const float* __restrict__ Wf,
                                                       const float* __restrict__ Wb,
                                                       bf16_t* __restrict__ Wt) {
  int i = blockIdx.x * 256 + threadIdx.x;
  if (i >= GATES * KPAD) return;
  int n = i / KPAD, k = i - n * KPAD;
  float v = 0.0f;
  if (k < EMB_K) v = (n < G4) ? Wf[(size_t)n * EMB_K + k] : Wb[(size_t)(n - G4) * EMB_K + k];
  Wt[i] = f2bf(v);
}

// ---------------------------------------------------------------------------
// K2: embedding gather + fp32->bf16, K padded to 320 with zeros
// ---------------------------------------------------------------------------
__global__ __launch_bounds__(256) void gather_embed_kernel(const int* __restrict__ idx,
                                                           const float* __restrict__ emb,
                                                           bf16_t* __restrict__ X, int rows) {
  size_t total = (size_t)rows * KPAD;
  size_t stride = (size_t)gridDim.x * blockDim.x;
  for (size_t i = (size_t)blockIdx.x * blockDim.x + threadIdx.x; i < total; i += stride) {
    int row = (int)(i / KPAD), k = (int)(i - (size_t)row * KPAD);
    float v = 0.0f;
    if (k < EMB_K) v = emb[(size_t)idx[row] * EMB_K + k];
    X[i] = f2bf(v);
  }
}

// ---------------------------------------------------------------------------
// K3: xproj GEMM  Y[M,512] = X[M,320] * Wt^T   (bf16 WMMA, fp32 acc -> bf16)
// block = 256 threads (8 waves); block tile 64(M) x 128(N); wave tile 16x64.
// ---------------------------------------------------------------------------
__global__ __launch_bounds__(256) void gemm_xproj_kernel(const bf16_t* __restrict__ X,
                                                         const bf16_t* __restrict__ Wt,
                                                         bf16_t* __restrict__ Y, int M) {
  int wave = threadIdx.x >> 5, lane = threadIdx.x & 31;
  int m0 = blockIdx.x * 64 + (wave >> 1) * 16;
  int n0 = blockIdx.y * 128 + (wave & 1) * 64;
  if (m0 >= M) return;
  v8f acc[4] = {};
  for (int k0 = 0; k0 < KPAD; k0 += 32) {
    const bf16_t* abase = X + (size_t)m0 * KPAD + k0;
    if (k0 + 32 < KPAD) __builtin_prefetch(abase + 32, 0, 0);  // global_prefetch_b8
    v16bf a = load_a_frag(abase, KPAD, lane);
#pragma unroll
    for (int nt = 0; nt < 4; ++nt) {
      v16bf b = load_bt_frag(Wt + (size_t)(n0 + nt * 16) * KPAD + k0, KPAD, lane);
      acc[nt] = WMMA_BF16(a, b, acc[nt]);
    }
  }
  int nl = lane & 15, mg = lane >> 4;
#pragma unroll
  for (int nt = 0; nt < 4; ++nt)
#pragma unroll
    for (int j = 0; j < 8; ++j) {
      size_t m = (size_t)(m0 + mg * 8 + j);
      Y[m * GATES + (n0 + nt * 16 + nl)] = f2bf(acc[nt][j]);
    }
}

// ---------------------------------------------------------------------------
// K4: masked LSTM scan (one direction per blockIdx.y, 16 batch rows / block).
// Per step: G = h @ Whh^T (WMMA in LDS) + xproj + bias; gate math; masked
// state update; fwd writes out[t], bwd writes out[len-1-t].
// Static LDS = 64,576 B (< 64 KiB).
// ---------------------------------------------------------------------------
__global__ __launch_bounds__(256) void lstm_scan_kernel(const bf16_t* __restrict__ xproj,
                                                        const float* __restrict__ WhhF,
                                                        const float* __restrict__ biasF,
                                                        const float* __restrict__ WhhB,
                                                        const float* __restrict__ biasB,
                                                        const int* __restrict__ lens,
                                                        float* __restrict__ outM, int T) {
  __shared__ bf16_t s_whh[G4 * HH];     // [n][k] 32 KB
  __shared__ float  s_bias[G4];
  __shared__ bf16_t s_h[16 * HH];
  __shared__ float  s_c[16 * HH];
  __shared__ float  s_g[16 * G4];       // 16 KB
  __shared__ bf16_t s_xp[16 * G4];      // 8 KB
  __shared__ int    s_len[16];

  int tid = threadIdx.x, wave = tid >> 5, lane = tid & 31;
  int bbase = blockIdx.x * 16;
  int dir = blockIdx.y;                 // 0 fwd, 1 bwd
  const float* Whh  = dir ? WhhB : WhhF;
  const float* bias = dir ? biasB : biasF;
  int gofs = dir ? G4 : 0;

  for (int i = tid; i < G4 * HH; i += 256) s_whh[i] = f2bf(Whh[i]);
  if (tid < G4) s_bias[tid] = bias[tid];
  for (int i = tid; i < 16 * HH; i += 256) { s_h[i] = f2bf(0.0f); s_c[i] = 0.0f; }
  if (tid < 16) s_len[tid] = lens[bbase + tid];
  __syncthreads();

  for (int t = 0; t < T; ++t) {
    // stage xproj rows for this step (16 rows x 256 gate cols, bf16)
    {
      int r = tid >> 4, cseg = (tid & 15) * 16;
      int len = s_len[r];
      int tt = dir ? (len - 1 - t) : t;
      if (tt < 0) tt = 0; if (tt >= T) tt = T - 1;         // clamped; masked anyway
      const bf16_t* src = xproj + ((size_t)(bbase + r) * T + tt) * GATES + gofs + cseg;
      *(bf16x8*)&s_xp[r * G4 + cseg]     = *(const bf16x8*)src;
      *(bf16x8*)&s_xp[r * G4 + cseg + 8] = *(const bf16x8*)(src + 8);
    }
    __syncthreads();
    // G[16,256] = h[16,64] @ Whh^T  (each wave: 2 n-tiles of 16)
    {
      int n0 = wave * 32;
      v8f a0 = {}, a1 = {};
#pragma unroll
      for (int k0 = 0; k0 < HH; k0 += 32) {
        v16bf a  = load_a_frag(s_h + k0, HH, lane);
        v16bf b0 = load_bt_frag(s_whh + (size_t)n0 * HH + k0, HH, lane);
        v16bf b1 = load_bt_frag(s_whh + (size_t)(n0 + 16) * HH + k0, HH, lane);
        a0 = WMMA_BF16(a, b0, a0);
        a1 = WMMA_BF16(a, b1, a1);
      }
      int nl = lane & 15, mg = lane >> 4;
#pragma unroll
      for (int j = 0; j < 8; ++j) {
        s_g[(mg * 8 + j) * G4 + n0 + nl]      = a0[j];
        s_g[(mg * 8 + j) * G4 + n0 + 16 + nl] = a1[j];
      }
    }
    __syncthreads();
    // gate math: 16 rows x 64 cells, 4 cells/thread
#pragma unroll
    for (int q = 0; q < 4; ++q) {
      int cell = tid + 256 * q;
      int r = cell >> 6, hc = cell & 63;
      int len = s_len[r];
      float gi = s_g[r * G4 + hc]        + bf2f(s_xp[r * G4 + hc])        + s_bias[hc];
      float gf = s_g[r * G4 + 64 + hc]   + bf2f(s_xp[r * G4 + 64 + hc])   + s_bias[64 + hc];
      float gg = s_g[r * G4 + 128 + hc]  + bf2f(s_xp[r * G4 + 128 + hc])  + s_bias[128 + hc];
      float go = s_g[r * G4 + 192 + hc]  + bf2f(s_xp[r * G4 + 192 + hc])  + s_bias[192 + hc];
      float iv = sigmoidf(gi), fv = sigmoidf(gf), gv = tanhf(gg), ov = sigmoidf(go);
      float cn = fv * s_c[r * HH + hc] + iv * gv;
      float hn = ov * tanhf(cn);
      if (t < len) {
        s_c[r * HH + hc] = cn;
        s_h[r * HH + hc] = f2bf(hn);
        int tout = dir ? (len - 1 - t) : t;
        outM[(((size_t)(bbase + r)) * T + tout) * DM + dir * HH + hc] = hn;
      }
    }
    __syncthreads();
  }
}

// ---------------------------------------------------------------------------
// K5: tvec[row] = M[row]·w ; Mbf[row] = bf16(M[row] * w3)  (w3==null -> copy)
// ---------------------------------------------------------------------------
__global__ __launch_bounds__(256) void prep_attn_kernel(const float* __restrict__ M,
                                                        const float* __restrict__ wdot,
                                                        const float* __restrict__ w3,
                                                        bf16_t* __restrict__ Mbf,
                                                        float* __restrict__ tvec, int rows) {
  int row = blockIdx.x * 256 + threadIdx.x;
  if (row >= rows) return;
  const float* mr = M + (size_t)row * DM;
  bf16_t* ob = Mbf + (size_t)row * DM;
  float acc = 0.0f;
#pragma unroll 4
  for (int d = 0; d < DM; ++d) {
    float v = mr[d];
    acc += v * wdot[d];
    ob[d] = f2bf(w3 ? v * w3[d] : v);
  }
  tvec[row] = acc;
}

// ---------------------------------------------------------------------------
// K6: alignment tile: S = (body∘w3) @ pun^T + t1 + t2, fused row/col max.
// grid (32 istrips, 128 b); each block owns 16 body rows x all 256 pun cols.
// ---------------------------------------------------------------------------
__global__ __launch_bounds__(256) void align_kernel(const bf16_t* __restrict__ bodyW3,
                                                    const bf16_t* __restrict__ punBf,
                                                    const float* __restrict__ t1,
                                                    const float* __restrict__ t2,
                                                    float* __restrict__ rmax,
                                                    float* __restrict__ colpart) {
  __shared__ bf16_t s_a[16 * DM];
  __shared__ float  s_al[16][TPUN + 1];
  __shared__ float  s_part[16][16];
  int tid = threadIdx.x, wave = tid >> 5, lane = tid & 31;
  int b = blockIdx.y, is = blockIdx.x, i0 = is * 16;

  for (int i = tid; i < 16 * DM; i += 256)
    s_a[i] = bodyW3[((size_t)b * TBODY + i0) * DM + i];
  __syncthreads();

#pragma unroll
  for (int jt = 0; jt < 2; ++jt) {
    int j0 = wave * 32 + jt * 16;
    v8f acc = {};
#pragma unroll
    for (int k0 = 0; k0 < DM; k0 += 32) {
      v16bf a = load_a_frag(s_a + k0, DM, lane);
      v16bf bb = load_bt_frag(punBf + ((size_t)b * TPUN + j0) * DM + k0, DM, lane);
      acc = WMMA_BF16(a, bb, acc);
    }
    int nl = lane & 15, mg = lane >> 4;
#pragma unroll
    for (int j = 0; j < 8; ++j) {
      int r = mg * 8 + j;
      s_al[r][j0 + nl] = acc[j] + t1[(size_t)b * TBODY + i0 + r] + t2[(size_t)b * TPUN + j0 + nl];
    }
  }
  __syncthreads();
  { // row max (exact: block covers all j)
    int row = tid >> 4, seg = (tid & 15) * 16;
    float m = -3.402823466e38f;
#pragma unroll
    for (int c = 0; c < 16; ++c) m = fmaxf(m, s_al[row][seg + c]);
    s_part[row][tid & 15] = m;
  }
  __syncthreads();
  if (tid < 16) {
    float m = -3.402823466e38f;
#pragma unroll
    for (int s = 0; s < 16; ++s) m = fmaxf(m, s_part[tid][s]);
    rmax[(size_t)b * TBODY + i0 + tid] = m;
  }
  { // partial col max over this istrip
    float m = -3.402823466e38f;
#pragma unroll
    for (int r = 0; r < 16; ++r) m = fmaxf(m, s_al[r][tid]);
    colpart[((size_t)b * 32 + is) * TPUN + tid] = m;
  }
}

__global__ __launch_bounds__(256) void col_reduce_kernel(const float* __restrict__ colpart,
                                                         float* __restrict__ cmax) {
  int idx = blockIdx.x * 256 + threadIdx.x;   // b*256 + j
  int b = idx >> 8, j = idx & 255;
  float m = -3.402823466e38f;
#pragma unroll 4
  for (int s = 0; s < 32; ++s) m = fmaxf(m, colpart[((size_t)b * 32 + s) * TPUN + j]);
  cmax[idx] = m;
}

// ---------------------------------------------------------------------------
// K8: softmax attentions + features + final linear (one block per batch row)
// ---------------------------------------------------------------------------
__device__ __forceinline__ float blk_max(float v, float* s) {
  int t = threadIdx.x;
  s[t] = v; __syncthreads();
  for (int o = 128; o > 0; o >>= 1) { if (t < o) s[t] = fmaxf(s[t], s[t + o]); __syncthreads(); }
  float r = s[0]; __syncthreads(); return r;
}
__device__ __forceinline__ float blk_sum(float v, float* s) {
  int t = threadIdx.x;
  s[t] = v; __syncthreads();
  for (int o = 128; o > 0; o >>= 1) { if (t < o) s[t] += s[t + o]; __syncthreads(); }
  float r = s[0]; __syncthreads(); return r;
}

__global__ __launch_bounds__(256) void attn_out_kernel(const float* __restrict__ bodyM,
                                                       const float* __restrict__ punM,
                                                       const float* __restrict__ rmax,
                                                       const float* __restrict__ cmax,
                                                       const float* __restrict__ Wd,
                                                       const float* __restrict__ bd,
                                                       float* __restrict__ out) {
  __shared__ float s_e[TBODY];
  __shared__ float s_ep[TPUN];
  __shared__ float s_red[256];
  __shared__ float s_feat[256];
  int b = blockIdx.x, tid = threadIdx.x;

  // body softmax over T (full axis, padded rows included — matches reference)
  float lm = -3.402823466e38f;
  for (int t = tid; t < TBODY; t += 256) lm = fmaxf(lm, rmax[(size_t)b * TBODY + t]);
  float mx = blk_max(lm, s_red);
  float ls = 0.0f;
  for (int t = tid; t < TBODY; t += 256) {
    float e = expf(rmax[(size_t)b * TBODY + t] - mx);
    s_e[t] = e; ls += e;
  }
  float Sb = blk_sum(ls, s_red);

  // pun softmax over P
  float v2 = cmax[(size_t)b * TPUN + tid];
  float mx2 = blk_max(v2, s_red);
  float e2 = expf(v2 - mx2);
  s_ep[tid] = e2;
  float Sp = blk_sum(e2, s_red);
  __syncthreads();

  if (tid < DM) {
    float acc = 0.0f;
    for (int t = 0; t < TBODY; ++t) acc += s_e[t] * bodyM[((size_t)b * TBODY + t) * DM + tid];
    s_feat[tid] = acc / Sb;
  } else {
    int d = tid - DM;
    float acc = 0.0f;
    for (int p = 0; p < TPUN; ++p) acc += s_ep[p] * punM[((size_t)b * TPUN + p) * DM + d];
    s_feat[tid] = acc / Sp;
  }
  __syncthreads();
  if (tid < 3) {
    float acc = bd[tid];
#pragma unroll 4
    for (int d = 0; d < 256; ++d) acc += s_feat[d] * Wd[tid * 256 + d];
    out[b * 3 + tid] = acc;
  }
}

// ---------------------------------------------------------------------------
// Host launcher. Inputs in setup_inputs() dict order:
// 0 body_idx 1 pun_idx 2 emb 3 w_u 4 Wd 5 bd
// 6 bWih_f 7 bWhh_f 8 bb_f 9 bWih_b 10 bWhh_b 11 bb_b
// 12 pWih_f 13 pWhh_f 14 pb_f 15 pWih_b 16 pWhh_b 17 pb_b
// ---------------------------------------------------------------------------
extern "C" void kernel_launch(void* const* d_in, const int* in_sizes, int n_in,
                              void* d_out, int out_size, void* d_ws, size_t ws_size,
                              hipStream_t stream) {
  (void)in_sizes; (void)n_in; (void)out_size; (void)ws_size;
  const int*   body_idx = (const int*)d_in[0];
  const int*   pun_idx  = (const int*)d_in[1];
  const float* emb      = (const float*)d_in[2];
  const float* w_u      = (const float*)d_in[3];
  const float* Wd       = (const float*)d_in[4];
  const float* bd       = (const float*)d_in[5];
  const float* bWih_f   = (const float*)d_in[6];
  const float* bWhh_f   = (const float*)d_in[7];
  const float* bb_f     = (const float*)d_in[8];
  const float* bWih_b   = (const float*)d_in[9];
  const float* bWhh_b   = (const float*)d_in[10];
  const float* bb_b     = (const float*)d_in[11];
  const float* pWih_f   = (const float*)d_in[12];
  const float* pWhh_f   = (const float*)d_in[13];
  const float* pb_f     = (const float*)d_in[14];
  const float* pWih_b   = (const float*)d_in[15];
  const float* pWhh_b   = (const float*)d_in[16];
  const float* pb_b     = (const float*)d_in[17];
  float* out = (float*)d_out;

  const int MB = NB * TBODY;   // 65536
  const int MP = NB * TPUN;    // 32768

  char* p = (char*)d_ws;
  auto take = [&](size_t n) { void* r = (void*)p; p += (n + 255) & ~(size_t)255; return r; };
  bf16_t* Xb      = (bf16_t*)take((size_t)MB * KPAD * 2);
  bf16_t* Xp      = (bf16_t*)take((size_t)MP * KPAD * 2);
  bf16_t* WtB     = (bf16_t*)take((size_t)GATES * KPAD * 2);
  bf16_t* WtP     = (bf16_t*)take((size_t)GATES * KPAD * 2);
  bf16_t* xprojB  = (bf16_t*)take((size_t)MB * GATES * 2);
  bf16_t* xprojP  = (bf16_t*)take((size_t)MP * GATES * 2);
  float*  bodyM   = (float*)take((size_t)MB * DM * 4);
  float*  punM    = (float*)take((size_t)MP * DM * 4);
  bf16_t* bodyW3  = (bf16_t*)take((size_t)MB * DM * 2);
  bf16_t* punBf   = (bf16_t*)take((size_t)MP * DM * 2);
  float*  t1      = (float*)take((size_t)MB * 4);
  float*  t2      = (float*)take((size_t)MP * 4);
  float*  rmaxB   = (float*)take((size_t)MB * 4);
  float*  colpart = (float*)take((size_t)NB * 32 * TPUN * 4);
  float*  cmaxB   = (float*)take((size_t)NB * TPUN * 4);
  int*    lens    = (int*)take((size_t)2 * NB * 4);

  // zero M buffers: masked/padded timesteps must read as exact zeros
  hipMemsetAsync(bodyM, 0, (size_t)MB * DM * 4, stream);
  hipMemsetAsync(punM,  0, (size_t)MP * DM * 4, stream);

  seq_lens_kernel<<<dim3(NB, 2), 256, 0, stream>>>(body_idx, pun_idx, lens);
  pack_wih_kernel<<<(GATES * KPAD + 255) / 256, 256, 0, stream>>>(bWih_f, bWih_b, WtB);
  pack_wih_kernel<<<(GATES * KPAD + 255) / 256, 256, 0, stream>>>(pWih_f, pWih_b, WtP);
  gather_embed_kernel<<<4096, 256, 0, stream>>>(body_idx, emb, Xb, MB);
  gather_embed_kernel<<<2048, 256, 0, stream>>>(pun_idx, emb, Xp, MP);
  gemm_xproj_kernel<<<dim3(MB / 64, GATES / 128), 256, 0, stream>>>(Xb, WtB, xprojB, MB);
  gemm_xproj_kernel<<<dim3(MP / 64, GATES / 128), 256, 0, stream>>>(Xp, WtP, xprojP, MP);
  lstm_scan_kernel<<<dim3(NB / 16, 2), 256, 0, stream>>>(xprojB, bWhh_f, bb_f, bWhh_b, bb_b,
                                                         lens, bodyM, TBODY);
  lstm_scan_kernel<<<dim3(NB / 16, 2), 256, 0, stream>>>(xprojP, pWhh_f, pb_f, pWhh_b, pb_b,
                                                         lens + NB, punM, TPUN);
  prep_attn_kernel<<<(MB + 255) / 256, 256, 0, stream>>>(bodyM, w_u + 0, w_u + 256, bodyW3, t1, MB);
  prep_attn_kernel<<<(MP + 255) / 256, 256, 0, stream>>>(punM, w_u + 128, nullptr, punBf, t2, MP);
  align_kernel<<<dim3(TBODY / 16, NB), 256, 0, stream>>>(bodyW3, punBf, t1, t2, rmaxB, colpart);
  col_reduce_kernel<<<(NB * TPUN) / 256, 256, 0, stream>>>(colpart, cmaxB);
  attn_out_kernel<<<NB, 256, 0, stream>>>(bodyM, punM, rmaxB, cmaxB, Wd, bd, out);
}
